// SymmetricLSTM_69071664054467
// MI455X (gfx1250) — compile-verified, matrix operand order checked by reading, not run on hardware
//
#include <hip/hip_runtime.h>
#include <hip/hip_bf16.h>
#include <math.h>

// B=64, S=512, D=512, H=512, 4H=2048
typedef __bf16 bf16;
typedef __attribute__((ext_vector_type(16))) __bf16 v16bf;
typedef __attribute__((ext_vector_type(8)))  float  v8f;

union ABfrag { v16bf v; uint4 q[2]; };

__device__ __forceinline__ float sigm_(float x) { return 1.0f / (1.0f + __expf(-x)); }
__device__ __forceinline__ float tanh_(float x) {
    float e = __expf(2.0f * x);
    return (e - 1.0f) / (e + 1.0f);
}

// ---------------------------------------------------------------------------
// fp32 -> bf16 conversion (grid-stride)
// ---------------------------------------------------------------------------
__global__ void k_cvt(const float* __restrict__ in, bf16* __restrict__ out, int n) {
    for (int i = blockIdx.x * blockDim.x + threadIdx.x; i < n; i += gridDim.x * blockDim.x)
        out[i] = (bf16)in[i];
}

// ---------------------------------------------------------------------------
// Phase 1: xg[m][n] = sum_k x_row(m)[k] * W_ih[n][k] + b_ih[n] + b_hh[n]
// GEMM row index m = s*64 + b  (so the scan reads contiguous slabs per step).
// Each wave computes a 64x64 tile; block = 8 waves -> 64(M) x 512(N).
// Grid: (32768/64, 2048/512) = (512, 4).
// ---------------------------------------------------------------------------
__global__ __launch_bounds__(256, 1) void k_gemm_xg(
    const bf16* __restrict__ xb, const bf16* __restrict__ wb,
    const float* __restrict__ bih, const float* __restrict__ bhh,
    bf16* __restrict__ xg)
{
    const int lane = threadIdx.x & 31;
    const int wv   = threadIdx.x >> 5;   // 0..7
    const int ln   = lane & 15;
    const int hi   = lane >> 4;          // 0/1
    const int kb   = hi << 3;            // K sub-offset per ISA A/B layout
    const int mbase = blockIdx.x * 64;
    const int nbase = blockIdx.y * 512 + wv * 64;

    const uint4* Ap[4];
    const uint4* Bp[4];
#pragma unroll
    for (int mi = 0; mi < 4; ++mi) {
        int m = mbase + mi * 16 + ln;        // m = s*64 + b
        int s = m >> 6, b = m & 63;
        Ap[mi] = (const uint4*)(xb + (size_t)(b * 512 + s) * 512);
    }
#pragma unroll
    for (int ni = 0; ni < 4; ++ni) {
        int r = nbase + ni * 16 + ln;        // W_ih row = output feature
        Bp[ni] = (const uint4*)(wb + (size_t)r * 512);
    }

    v8f c[4][4];
#pragma unroll
    for (int mi = 0; mi < 4; ++mi)
#pragma unroll
        for (int ni = 0; ni < 4; ++ni)
#pragma unroll
            for (int j = 0; j < 8; ++j) c[mi][ni][j] = 0.0f;

    for (int k0 = 0; k0 < 512; k0 += 32) {
        const int q0 = (k0 + kb) >> 3;       // uint4 index (8 bf16 per uint4)
        ABfrag a[4], bfr[4];
#pragma unroll
        for (int mi = 0; mi < 4; ++mi) { a[mi].q[0] = Ap[mi][q0]; a[mi].q[1] = Ap[mi][q0 + 2]; }
#pragma unroll
        for (int ni = 0; ni < 4; ++ni) { bfr[ni].q[0] = Bp[ni][q0]; bfr[ni].q[1] = Bp[ni][q0 + 2]; }
#pragma unroll
        for (int mi = 0; mi < 4; ++mi)
#pragma unroll
            for (int ni = 0; ni < 4; ++ni)
                c[mi][ni] = __builtin_amdgcn_wmma_f32_16x16x32_bf16(
                    false, a[mi].v, false, bfr[ni].v, (short)0, c[mi][ni], false, false);
    }

#pragma unroll
    for (int ni = 0; ni < 4; ++ni) {
        int n = nbase + ni * 16 + ln;
        float bias = bih[n] + bhh[n];
#pragma unroll
        for (int mi = 0; mi < 4; ++mi)
#pragma unroll
            for (int j = 0; j < 8; ++j) {
                int m = mbase + mi * 16 + hi * 8 + j;
                xg[(size_t)m * 2048 + n] = (bf16)(c[mi][ni][j] + bias);
            }
    }
}

// ---------------------------------------------------------------------------
// Phase 2: persistent recurrent scan. 4 blocks x 1024 threads.
// Block = 16 batch rows; wave wv owns h/c columns [16*wv, 16*wv+16).
// c-state lives in registers; h shared via LDS bf16 (pitch 520 to dodge
// bank conflicts on 16-byte fragment loads). No inter-block dependencies.
// Next timestep's contiguous 64KB xg slab is prefetched (global_prefetch_b8)
// during the current step's WMMA loop to take HBM latency off the serial
// recurrence chain.
// ---------------------------------------------------------------------------
__global__ __launch_bounds__(1024, 1) void k_scan(
    const bf16* __restrict__ xg, const bf16* __restrict__ whh,
    const float* __restrict__ gmh, const float* __restrict__ bth,
    const float* __restrict__ gmc, const float* __restrict__ btc,
    float* __restrict__ out)
{
    __shared__ bf16  sH[16 * 520];
    __shared__ float sRed[4][16];   // sum_c, ssq_c, sum_h, ssq_h per batch row

    const int lane = threadIdx.x & 31;
    const int wv   = threadIdx.x >> 5;   // 0..31
    const int ln   = lane & 15;
    const int hi   = lane >> 4;
    const int kb   = hi << 3;
    const int b0   = blockIdx.x * 16;
    const int n    = wv * 16 + ln;       // h/c column this lane owns

    for (int i = threadIdx.x; i < 16 * 520; i += blockDim.x) sH[i] = (bf16)0.0f;
    if (threadIdx.x < 64) ((float*)sRed)[threadIdx.x] = 0.0f;

    const float g_h = gmh[n], o_h = bth[n];
    const float g_c = gmc[n], o_c = btc[n];

    const uint4* Bp[4];
#pragma unroll
    for (int g = 0; g < 4; ++g)
        Bp[g] = (const uint4*)(whh + (size_t)(g * 512 + n) * 512);
    const uint4* Ap = (const uint4*)sH;
    const int arow = ln * 65;            // 520 bf16 = 65 uint4 per LDS row

    v8f cstate;
#pragma unroll
    for (int j = 0; j < 8; ++j) cstate[j] = 0.0f;

    // warm the cache for the first slab
    {
        const char* p0 = (const char*)(xg + (size_t)b0 * 2048);
        if (threadIdx.x < 512) __builtin_prefetch(p0 + threadIdx.x * 128, 0, 3);
    }

    __syncthreads();

    for (int t = 0; t < 512; ++t) {
        // gate accumulators, seeded with precomputed input projection
        v8f acc[4];
        const bf16* xgt = xg + (size_t)t * (64 * 2048);
#pragma unroll
        for (int g = 0; g < 4; ++g)
#pragma unroll
            for (int j = 0; j < 8; ++j) {
                int r = b0 + hi * 8 + j;
                acc[g][j] = (float)xgt[(size_t)r * 2048 + g * 512 + n];
            }

        // prefetch next step's 64KB slab (rows b0..b0+15 are contiguous)
        if (t + 1 < 512 && threadIdx.x < 512) {
            const char* nxt = (const char*)(xg + ((size_t)(t + 1) * 64 + b0) * 2048);
            __builtin_prefetch(nxt + threadIdx.x * 128, 0, 3);
        }

        // gates += h @ W_hh^T   (A from LDS, B from L2-resident W_hh)
        for (int k0 = 0; k0 < 512; k0 += 32) {
            const int q0 = (k0 + kb) >> 3;
            ABfrag a;
            a.q[0] = Ap[arow + q0];
            a.q[1] = Ap[arow + q0 + 2];
            ABfrag bb[4];
#pragma unroll
            for (int g = 0; g < 4; ++g) { bb[g].q[0] = Bp[g][q0]; bb[g].q[1] = Bp[g][q0 + 2]; }
#pragma unroll
            for (int g = 0; g < 4; ++g)
                acc[g] = __builtin_amdgcn_wmma_f32_16x16x32_bf16(
                    false, a.v, false, bb[g].v, (short)0, acc[g], false, false);
        }

        // LSTM cell elementwise (torch gate order: i, f, g, o)
        float cn[8], hp[8], s0[8], s1[8], s2[8], s3[8];
#pragma unroll
        for (int j = 0; j < 8; ++j) {
            float iv = sigm_(acc[0][j]);
            float fv = sigm_(acc[1][j]);
            float gv = tanh_(acc[2][j]);
            float ov = sigm_(acc[3][j]);
            float cv = fv * cstate[j] + iv * gv;
            float hv = ov * tanh_(cv);
            cn[j] = cv; hp[j] = hv;
            s0[j] = cv; s1[j] = cv * cv; s2[j] = hv; s3[j] = hv * hv;
        }

        // reduce over this wave's 16 columns (within each 16-lane half)
#pragma unroll
        for (int m = 1; m < 16; m <<= 1)
#pragma unroll
            for (int j = 0; j < 8; ++j) {
                s0[j] += __shfl_xor(s0[j], m, 16);
                s1[j] += __shfl_xor(s1[j], m, 16);
                s2[j] += __shfl_xor(s2[j], m, 16);
                s3[j] += __shfl_xor(s3[j], m, 16);
            }
        if (ln == 0) {
            int rb = hi * 8;
#pragma unroll
            for (int j = 0; j < 8; ++j) {
                atomicAdd(&sRed[0][rb + j], s0[j]);
                atomicAdd(&sRed[1][rb + j], s1[j]);
                atomicAdd(&sRed[2][rb + j], s2[j]);
                atomicAdd(&sRed[3][rb + j], s3[j]);
            }
        }
        __syncthreads();

        // symmetric layernorm of c (register carry) and h (LDS bf16 + output)
#pragma unroll
        for (int j = 0; j < 8; ++j) {
            int r = hi * 8 + j;
            float muc = sRed[0][r] * (1.0f / 512.0f);
            float vc  = sRed[1][r] * (1.0f / 512.0f) - muc * muc;
            cstate[j] = (cn[j] - muc) * rsqrtf(vc + 1e-5f) * g_c + o_c;
            float muh = sRed[2][r] * (1.0f / 512.0f);
            float vh  = sRed[3][r] * (1.0f / 512.0f) - muh * muh;
            float hvn = (hp[j] - muh) * rsqrtf(vh + 1e-5f) * g_h + o_h;
            out[((size_t)(b0 + r) * 512 + t) * 512 + n] = hvn;
            sH[r * 520 + n] = (bf16)hvn;
        }
        __syncthreads();
        if (threadIdx.x < 64) ((float*)sRed)[threadIdx.x] = 0.0f;
        __syncthreads();
    }
}

// ---------------------------------------------------------------------------
extern "C" void kernel_launch(void* const* d_in, const int* in_sizes, int n_in,
                              void* d_out, int out_size, void* d_ws, size_t ws_size,
                              hipStream_t stream)
{
    const float* x   = (const float*)d_in[0];
    const float* Wih = (const float*)d_in[1];
    const float* Whh = (const float*)d_in[2];
    const float* bih = (const float*)d_in[3];
    const float* bhh = (const float*)d_in[4];
    const float* gmh = (const float*)d_in[5];
    const float* bth = (const float*)d_in[6];
    const float* gmc = (const float*)d_in[7];
    const float* btc = (const float*)d_in[8];
    float* out = (float*)d_out;

    char* ws = (char*)d_ws;
    bf16* xb  = (bf16*)(ws);                                   // 33,554,432 B
    bf16* wib = (bf16*)(ws + 33554432);                        //  2,097,152 B
    bf16* whb = (bf16*)(ws + 33554432 + 2097152);              //  2,097,152 B
    bf16* xg  = (bf16*)(ws + 33554432 + 2 * 2097152);          // 134,217,728 B

    k_cvt<<<4096, 256, 0, stream>>>(x,   xb,  64 * 512 * 512);
    k_cvt<<<1024, 256, 0, stream>>>(Wih, wib, 2048 * 512);
    k_cvt<<<1024, 256, 0, stream>>>(Whh, whb, 2048 * 512);
    k_gemm_xg<<<dim3(512, 4), 256, 0, stream>>>(xb, wib, bih, bhh, xg);
    k_scan<<<4, 1024, 0, stream>>>(xg, whb, gmh, bth, gmc, btc, out);
}
